// ClipNet_58445914964557
// MI455X (gfx1250) — matrix operand order repaired and checked by reading, not compiled, not run on hardware
//
#include <hip/hip_runtime.h>

#define B_SZ     128
#define N_OTH    2048
#define F_IMG_D  1024
#define F_TXT_D  512
#define D_EMB    512
#define OUT_COLS 128           // TOPK + 1
#define TILE_ROWS 64           // rows per workgroup (4 waves x 16)
#define LDS_STRIDE 520         // 512 + 8 ushort pad -> conflict-free DS b128 reads
#define WBUF_USHORTS (16 * LDS_STRIDE)

typedef __attribute__((ext_vector_type(16))) __bf16 v16bf;
typedef __attribute__((ext_vector_type(8)))  float  v8f;

// Tensor DMA descriptor groups (CDNA5 ISA)
typedef unsigned int tdm_g0_t __attribute__((ext_vector_type(4)));
typedef int          tdm_g1_t __attribute__((ext_vector_type(8)));
typedef int          tdm_g2_t __attribute__((ext_vector_type(4)));

#if defined(__has_builtin)
#if __has_builtin(__builtin_amdgcn_tensor_load_to_lds) && __has_builtin(__builtin_amdgcn_s_wait_tensorcnt)
#define HAVE_TDM 1
#endif
#endif

#if defined(__has_include)
#if __has_include(<hip/amd_detail/amd_gfx1250_TDM.h>)
#define TDM_6ARG 1   // therock-10.0 headers -> clang-23 6-arg builtin
#endif
#endif

static __device__ __forceinline__ unsigned short f32_to_bf16_rne(float f) {
  unsigned int u = __float_as_uint(f);
  u += 0x7FFFu + ((u >> 16) & 1u);   // round to nearest even
  return (unsigned short)(u >> 16);
}

#if defined(HAVE_TDM)
// Issue one TDM load: 2D tile (512 bf16 x 16 rows) from Wt panel -> LDS,
// hardware-padded 16B per 1024B row so rows land LDS_STRIDE apart.
static __device__ __forceinline__ void tdm_load_panel(const unsigned short* gsrc,
                                                      unsigned int lds_byte_addr) {
  unsigned long long ga = (unsigned long long)(size_t)gsrc;
  tdm_g0_t g0;
  g0[0] = 1u;                                        // count=1, user mode
  g0[1] = lds_byte_addr;                             // lds_addr
  g0[2] = (unsigned int)(ga & 0xFFFFFFFFull);        // global_addr[31:0]
  g0[3] = (unsigned int)((ga >> 32) & 0x01FFFFFFull) // global_addr[56:32]
          | (2u << 30);                              // type = 2 ("image")
  tdm_g1_t g1;
  g1[0] = (int)((1u << 16)        // data_size = 2 bytes
              | (1u << 20)        // pad_enable
              | (7u << 22)        // pad_interval: 256 dwords (1024 B)
              | (3u << 25));      // pad_amount: 4 dwords (16 B)
  g1[1] = (int)(512u << 16);      // tensor_dim0[15:0]=512 (abar_addr=0)
  g1[2] = (int)(512u << 16);      // tensor_dim1[15:0]=512 (dim0 hi=0)
  g1[3] = (int)(512u << 16);      // tile_dim0=512          (dim1 hi=0)
  g1[4] = 16;                     // tile_dim1=16, tile_dim2=0
  g1[5] = 512;                    // tensor_dim0_stride[31:0]=512
  g1[6] = 0;                      // dim0_stride hi, dim1_stride lo
  g1[7] = 0;
  tdm_g2_t gz = {0, 0, 0, 0};
#if defined(TDM_6ARG)
  tdm_g1_t gz8 = {0, 0, 0, 0, 0, 0, 0, 0};
  __builtin_amdgcn_tensor_load_to_lds(g0, g1, gz, gz, gz8, 0);
#else
  __builtin_amdgcn_tensor_load_to_lds(g0, g1, gz, gz, 0);
#endif
}
#endif

// ---------------- W_txt f32 -> bf16, transposed: Wt[n][k] ----------------
__global__ void k_wt_cvt(const float* __restrict__ W, unsigned short* __restrict__ Wt) {
  int idx = blockIdx.x * 256 + threadIdx.x;       // 0 .. 512*512-1
  int k = idx >> 9;
  int n = idx & 511;
  Wt[n * D_EMB + k] = f32_to_bf16_rne(W[idx]);
}

// ---------------- image/text encoders + logit_in (tiny) ----------------
__global__ __launch_bounds__(256) void k_prep(
    const float* __restrict__ in_img, const float* __restrict__ in_txt,
    const float* __restrict__ W_img,  const float* __restrict__ W_txt,
    const float* __restrict__ p_ls,
    float* __restrict__ img_n, float* __restrict__ logit_in) {
  __shared__ float xi[F_IMG_D];
  __shared__ float rowi[D_EMB];
  __shared__ float rowt[D_EMB];
  __shared__ float red[256];
  const int b = blockIdx.x, tid = threadIdx.x;

  for (int i = tid; i < F_IMG_D; i += 256) xi[i] = in_img[b * F_IMG_D + i];
  __syncthreads();
  for (int d = tid; d < D_EMB; d += 256) {
    float acc = 0.f;
    for (int k = 0; k < F_IMG_D; ++k) acc = fmaf(xi[k], W_img[k * D_EMB + d], acc);
    rowi[d] = acc;
  }
  __syncthreads();
  red[tid] = rowi[tid] * rowi[tid] + rowi[tid + 256] * rowi[tid + 256];
  __syncthreads();
  for (int s = 128; s > 0; s >>= 1) { if (tid < s) red[tid] += red[tid + s]; __syncthreads(); }
  float rn = rsqrtf(red[0]);
  __syncthreads();
  for (int d = tid; d < D_EMB; d += 256) { rowi[d] *= rn; img_n[b * D_EMB + d] = rowi[d]; }
  __syncthreads();

  for (int i = tid; i < F_TXT_D; i += 256) xi[i] = in_txt[b * F_TXT_D + i];
  __syncthreads();
  for (int d = tid; d < D_EMB; d += 256) {
    float acc = 0.f;
    for (int k = 0; k < F_TXT_D; ++k) acc = fmaf(xi[k], W_txt[k * D_EMB + d], acc);
    rowt[d] = acc;
  }
  __syncthreads();
  red[tid] = rowt[tid] * rowt[tid] + rowt[tid + 256] * rowt[tid + 256];
  __syncthreads();
  for (int s = 128; s > 0; s >>= 1) { if (tid < s) red[tid] += red[tid + s]; __syncthreads(); }
  float rn2 = rsqrtf(red[0]);
  __syncthreads();
  red[tid] = rowi[tid] * rowt[tid] + rowi[tid + 256] * rowt[tid + 256];
  __syncthreads();
  for (int s = 128; s > 0; s >>= 1) { if (tid < s) red[tid] += red[tid + s]; __syncthreads(); }
  if (tid == 0) logit_in[b] = expf(p_ls[0]) * red[0] * rn2;
}

// ---------------- fused projection + norm + dot  (WMMA bf16 + TDM) ----------------
__global__ __launch_bounds__(128) void k_logits(
    const float* __restrict__ other,          // [B, N, 512] f32
    const unsigned short* __restrict__ Wt,    // [512 n][512 k] bf16
    const float* __restrict__ img_n,          // [B, 512]
    const float* __restrict__ p_ls,
    float* __restrict__ logit_oth) {          // [B, N]
  extern __shared__ __align__(16) unsigned short smem[];
  unsigned short* lds_a = smem;                               // 64 x LDS_STRIDE
  unsigned short* lds_w = smem + TILE_ROWS * LDS_STRIDE;      // 2 x (16 x LDS_STRIDE)

  const int tid   = threadIdx.x;
  const int wave  = tid >> 5;                 // 0..3
  const int lane  = tid & 31;
  const int laneN = lane & 15;
  const int hi    = (lane >> 4) & 1;
  const int hi8   = hi * 8;

  const int b    = blockIdx.x >> 5;           // 32 tiles per batch
  const int tile = blockIdx.x & 31;
  const int row0 = tile * TILE_ROWS;

  const float scale = expf(p_ls[0]);

#if defined(HAVE_TDM)
  if (wave == 0)                               // prefetch W panel j=0 into buf0
    tdm_load_panel(Wt, (unsigned int)(size_t)(void*)lds_w);
#endif

  // ---- stage A tile: 64 rows x 512 f32 -> bf16 in LDS (HBM stream) ----
  {
    const float4* src = (const float4*)(other + ((size_t)b * N_OTH + row0) * (size_t)D_EMB);
    #pragma unroll 4
    for (int i = 0; i < 64; ++i) {
      int idx4 = i * 128 + tid;            // coalesced float4 index in tile
      int r    = idx4 >> 7;                // 128 float4 per row
      int c4   = idx4 & 127;
      float4 v = src[idx4];
      uint2 pk;
      pk.x = (unsigned int)f32_to_bf16_rne(v.x) | ((unsigned int)f32_to_bf16_rne(v.y) << 16);
      pk.y = (unsigned int)f32_to_bf16_rne(v.z) | ((unsigned int)f32_to_bf16_rne(v.w) << 16);
      *(uint2*)(lds_a + r * LDS_STRIDE + c4 * 4) = pk;
    }
  }

  float sumsq[8], dotv[8];
  #pragma unroll
  for (int r = 0; r < 8; ++r) { sumsq[r] = 0.f; dotv[r] = 0.f; }

  const unsigned short* a_base = lds_a + (wave * 16 + laneN) * LDS_STRIDE;

  union Frag { uint4 q[2]; v16bf v; };

  for (int j = 0; j < 32; ++j) {              // 32 output-column tiles of 16
    const int p = j & 1;
#if defined(HAVE_TDM)
    if (wave == 0) __builtin_amdgcn_s_wait_tensorcnt(0);  // panel j landed
    __syncthreads();                          // panel j visible; buf 1-p free
    if (wave == 0 && j < 31)                  // async prefetch panel j+1
      tdm_load_panel(Wt + (size_t)(j + 1) * 16 * D_EMB,
                     (unsigned int)(size_t)(void*)(lds_w + (1 - p) * WBUF_USHORTS));
#else
    __syncthreads();                          // lds_w free to overwrite
    {                                         // manual panel copy fallback
      int r = tid >> 3, seg = tid & 7;        // 16 rows x 8 segs of 64 ushorts
      const uint4* s4 = (const uint4*)(Wt + (size_t)(j * 16 + r) * D_EMB + seg * 64);
      uint4* d4 = (uint4*)(lds_w + p * WBUF_USHORTS + r * LDS_STRIDE + seg * 64);
      #pragma unroll
      for (int q = 0; q < 8; ++q) d4[q] = s4[q];
    }
    __syncthreads();
#endif

    const unsigned short* wb = lds_w + p * WBUF_USHORTS + laneN * LDS_STRIDE;

    v8f c;
    #pragma unroll
    for (int r = 0; r < 8; ++r) c[r] = 0.f;

    // software-pipelined K loop: preload kt+1 fragments, then WMMA kt
    Frag fa, fb;
    {
      const uint4* pa = (const uint4*)(a_base + hi * 16);
      fa.q[0] = pa[0]; fa.q[1] = pa[1];
      const uint4* pb = (const uint4*)(wb + hi * 16);
      fb.q[0] = pb[0]; fb.q[1] = pb[1];
    }
    #pragma unroll
    for (int kt = 0; kt < 16; ++kt) {
      Frag na = fa, nb = fb;
      if (kt < 15) {
        const uint4* pa = (const uint4*)(a_base + (kt + 1) * 32 + hi * 16);
        na.q[0] = pa[0]; na.q[1] = pa[1];
        const uint4* pb = (const uint4*)(wb + (kt + 1) * 32 + hi * 16);
        nb.q[0] = pb[0]; nb.q[1] = pb[1];
      }
      c = __builtin_amdgcn_wmma_f32_16x16x32_bf16(false, fa.v, false, fb.v,
                                                  (short)0, c, false, false);
      fa = na; fb = nb;
    }

    const float imv = img_n[b * D_EMB + j * 16 + laneN];   // column N = laneN
    #pragma unroll
    for (int r = 0; r < 8; ++r) {             // c[r] -> row M = hi*8 + r, col N = laneN
      float s = c[r];
      sumsq[r] = fmaf(s, s, sumsq[r]);
      dotv[r]  = fmaf(s, imv, dotv[r]);
    }
  }

  // reduce over the 16 lanes of each half (masks 1,2,4,8 keep halves separate)
  #pragma unroll
  for (int r = 0; r < 8; ++r) {
    #pragma unroll
    for (int m = 1; m <= 8; m <<= 1) {
      sumsq[r] += __shfl_xor(sumsq[r], m, 32);
      dotv[r]  += __shfl_xor(dotv[r],  m, 32);
    }
  }

  if (laneN == 0) {
    int rbase = row0 + wave * 16 + hi8;
    #pragma unroll
    for (int r = 0; r < 8; ++r)
      logit_oth[(size_t)b * N_OTH + rbase + r] = scale * dotv[r] * rsqrtf(sumsq[r]);
  }
}

// ---------------- per-row bitonic sort (desc) + output assembly ----------------
__global__ __launch_bounds__(256) void k_topk(
    const float* __restrict__ logit_oth, const float* __restrict__ logit_in,
    float* __restrict__ out) {
  __shared__ float s[N_OTH];
  const int b = blockIdx.x, tid = threadIdx.x;
  for (int i = tid; i < N_OTH; i += 256) s[i] = logit_oth[(size_t)b * N_OTH + i];
  __syncthreads();
  for (int k = 2; k <= N_OTH; k <<= 1) {
    for (int jj = k >> 1; jj > 0; jj >>= 1) {
      for (int i = tid; i < N_OTH; i += 256) {
        int ixj = i ^ jj;
        if (ixj > i) {
          float a = s[i], c = s[ixj];
          bool desc = ((i & k) == 0);
          if (desc ? (a < c) : (a > c)) { s[i] = c; s[ixj] = a; }
        }
      }
      __syncthreads();
    }
  }
  if (tid < OUT_COLS) {
    int c = tid;
    int p = (b < OUT_COLS - 2) ? b : (OUT_COLS - 2);  // min(b, K-1)
    int src = (c <= p) ? c : c - 1;
    float v = (c == b) ? logit_in[b] : s[src];
    out[b * OUT_COLS + c] = v;
  }
}

extern "C" void kernel_launch(void* const* d_in, const int* in_sizes, int n_in,
                              void* d_out, int out_size, void* d_ws, size_t ws_size,
                              hipStream_t stream) {
  const float* in_img = (const float*)d_in[0];
  const float* in_txt = (const float*)d_in[1];
  const float* other  = (const float*)d_in[2];
  const float* W_img  = (const float*)d_in[3];
  const float* W_txt  = (const float*)d_in[4];
  const float* p_ls   = (const float*)d_in[5];
  float* out = (float*)d_out;

  // workspace layout (all 16B-aligned offsets)
  float* img_n     = (float*)d_ws;                        // 128*512 f32
  float* logit_in  = img_n + B_SZ * D_EMB;                // 128 f32
  unsigned short* Wt = (unsigned short*)(logit_in + B_SZ);// 512*512 bf16
  float* logit_oth = (float*)(Wt + D_EMB * D_EMB);        // 128*2048 f32

  k_wt_cvt<<<(D_EMB * D_EMB) / 256, 256, 0, stream>>>(W_txt, Wt);
  k_prep<<<B_SZ, 256, 0, stream>>>(in_img, in_txt, W_img, W_txt, p_ls, img_n, logit_in);

  size_t smem = (size_t)(TILE_ROWS * LDS_STRIDE + 2 * WBUF_USHORTS) * sizeof(unsigned short);
  hipFuncSetAttribute(reinterpret_cast<const void*>(k_logits),
                      hipFuncAttributeMaxDynamicSharedMemorySize, (int)smem);
  k_logits<<<B_SZ * (N_OTH / TILE_ROWS), 128, smem, stream>>>(other, Wt, img_n, p_ls, logit_oth);

  k_topk<<<B_SZ, 256, 0, stream>>>(logit_oth, logit_in, out);

  (void)in_sizes; (void)n_in; (void)out_size; (void)ws_size;
}
